// MultiscaleExtractor_36438502539816
// MI455X (gfx1250) — matrix operand order repaired and checked by reading, not compiled, not run on hardware
//
#include <hip/hip_runtime.h>

typedef _Float16 half8  __attribute__((ext_vector_type(8)));
typedef _Float16 v16h   __attribute__((ext_vector_type(16)));
typedef _Float16 half4v __attribute__((ext_vector_type(4)));
typedef float    v8f    __attribute__((ext_vector_type(8)));

union V16 { half8 h[2]; v16h v; };

#define LQ    21504
#define LIN   4096
#define CDIM  768
#define NHEAD 6
#define DHEAD 128
#define HIDF  192
#define IMG_H 64
#define IMG_W 64

// ---------------- LayerNorm: f32 row (768) -> f16 row ----------------
__global__ __launch_bounds__(256) void ln_kernel(const float* __restrict__ x,
    const float* __restrict__ g, const float* __restrict__ b,
    _Float16* __restrict__ out)
{
  __shared__ float s1[256], s2[256];
  int row = blockIdx.x;
  const float* xr = x + (size_t)row * CDIM;
  int t = threadIdx.x;
  float v0 = xr[t], v1 = xr[t + 256], v2 = xr[t + 512];
  s1[t] = v0 + v1 + v2;
  s2[t] = v0 * v0 + v1 * v1 + v2 * v2;
  __syncthreads();
  for (int st = 128; st > 0; st >>= 1) {
    if (t < st) { s1[t] += s1[t + st]; s2[t] += s2[t + st]; }
    __syncthreads();
  }
  float mu  = s1[0] * (1.0f / CDIM);
  float var = s2[0] * (1.0f / CDIM) - mu * mu;
  float rs  = rsqrtf(var + 1e-6f);
  _Float16* orow = out + (size_t)row * CDIM;
  orow[t]       = (_Float16)((v0 - mu) * rs * g[t]       + b[t]);
  orow[t + 256] = (_Float16)((v1 - mu) * rs * g[t + 256] + b[t + 256]);
  orow[t + 512] = (_Float16)((v2 - mu) * rs * g[t + 512] + b[t + 512]);
}

// ---- WMMA GEMM, 32x32 tile per wave (2x2 register blocking, 4 WMMA / 8 loads) ----
// C[M,N] = A[M,K](f16, row-major) * Bt[N,K](f16, row-major = B^T) + bias (+resid)
__global__ __launch_bounds__(256) void gemm_wmma(const _Float16* __restrict__ A,
    const _Float16* __restrict__ Bt, const float* __restrict__ bias,
    const float* __restrict__ resid, float* __restrict__ C, int N, int K)
{
  int lane = threadIdx.x & 31;
  int wave = threadIdx.x >> 5;
  int nm   = lane & 15;
  int sel  = lane >> 4;                       // 0: low half, 1: high half
  int m0 = (blockIdx.y * 8 + wave) * 32;      // 8 waves stacked in M: 256-row block
  int n0 = blockIdx.x * 32;
  const _Float16* arow0 = A  + (size_t)(m0 + nm) * K;
  const _Float16* arow1 = A  + (size_t)(m0 + 16 + nm) * K;
  const _Float16* brow0 = Bt + (size_t)(n0 + nm) * K;
  const _Float16* brow1 = Bt + (size_t)(n0 + 16 + nm) * K;
  float bv0 = bias ? bias[n0 + nm]      : 0.0f;
  float bv1 = bias ? bias[n0 + 16 + nm] : 0.0f;
  v8f c00, c01, c10, c11;
#pragma unroll
  for (int i = 0; i < 8; i++) { c00[i] = bv0; c01[i] = bv1; c10[i] = bv0; c11[i] = bv1; }
  int aoff = sel * 8;                         // A: K = sel*8 + [0,8) and 16+sel*8+[0,8)
  int boff = sel * 8;                         // B: K = sel*8 + [0,8) and 16+sel*8+[0,8)
  for (int kk = 0; kk < K; kk += 32) {
    V16 a0, a1, b0, b1;
    a0.h[0] = *(const half8*)(arow0 + kk + aoff);
    a0.h[1] = *(const half8*)(arow0 + kk + 16 + aoff);
    a1.h[0] = *(const half8*)(arow1 + kk + aoff);
    a1.h[1] = *(const half8*)(arow1 + kk + 16 + aoff);
    b0.h[0] = *(const half8*)(brow0 + kk + boff);
    b0.h[1] = *(const half8*)(brow0 + kk + boff + 16);
    b1.h[0] = *(const half8*)(brow1 + kk + boff);
    b1.h[1] = *(const half8*)(brow1 + kk + boff + 16);
    c00 = __builtin_amdgcn_wmma_f32_16x16x32_f16(false, a0.v, false, b0.v, (short)0, c00, false, false);
    c01 = __builtin_amdgcn_wmma_f32_16x16x32_f16(false, a0.v, false, b1.v, (short)0, c01, false, false);
    c10 = __builtin_amdgcn_wmma_f32_16x16x32_f16(false, a1.v, false, b0.v, (short)0, c10, false, false);
    c11 = __builtin_amdgcn_wmma_f32_16x16x32_f16(false, a1.v, false, b1.v, (short)0, c11, false, false);
  }
  int col0 = n0 + nm, col1 = n0 + 16 + nm;
#pragma unroll
  for (int i = 0; i < 8; i++) {
    int row0 = m0 + i + sel * 8;
    int row1 = row0 + 16;
    float v00 = c00[i], v01 = c01[i], v10 = c10[i], v11 = c11[i];
    if (resid) {
      v00 += resid[(size_t)row0 * N + col0];
      v01 += resid[(size_t)row0 * N + col1];
      v10 += resid[(size_t)row1 * N + col0];
      v11 += resid[(size_t)row1 * N + col1];
    }
    C[(size_t)row0 * N + col0] = v00;
    C[(size_t)row0 * N + col1] = v01;
    C[(size_t)row1 * N + col0] = v10;
    C[(size_t)row1 * N + col1] = v11;
  }
}

// ---------------- Deformable sampling: one wave per (query, head) ----------------
__global__ __launch_bounds__(256) void sample_kernel(
    const float* __restrict__ refp, const float* __restrict__ off,
    const float* __restrict__ awl, const float* __restrict__ value,
    _Float16* __restrict__ attn_s)
{
  int wid  = blockIdx.x * 8 + (threadIdx.x >> 5);
  int lane = threadIdx.x & 31;
  int q = wid / NHEAD;
  int m = wid - q * NHEAD;
  float rx = refp[q * 2 + 0] * (float)IMG_W;
  float ry = refp[q * 2 + 1] * (float)IMG_H;
  const float* lg = awl + (size_t)q * 32 + m * 4;      // softmax over L*P=4
  float l0 = lg[0], l1 = lg[1], l2 = lg[2], l3 = lg[3];
  float mx = fmaxf(fmaxf(l0, l1), fmaxf(l2, l3));
  float e0 = expf(l0 - mx), e1 = expf(l1 - mx), e2 = expf(l2 - mx), e3 = expf(l3 - mx);
  float inv = 1.0f / (e0 + e1 + e2 + e3);
  float aw[4] = { e0 * inv, e1 * inv, e2 * inv, e3 * inv };
  int d0 = lane * 4;
  float a0 = 0.f, a1 = 0.f, a2 = 0.f, a3 = 0.f;
  const float* offq = off + (size_t)q * 96 + m * 8;    // (M,L,P,2)
  for (int p = 0; p < 4; p++) {
    float fx = rx + offq[p * 2 + 0] - 0.5f;            // ref*W + off - 0.5
    float fy = ry + offq[p * 2 + 1] - 0.5f;
    float x0f = floorf(fx), y0f = floorf(fy);
    float wx = fx - x0f, wy = fy - y0f;
    int x0 = (int)x0f, y0 = (int)y0f;
    float wgt[4] = { (1.f - wx) * (1.f - wy), wx * (1.f - wy),
                     (1.f - wx) * wy,         wx * wy };
    int xs[4] = { x0, x0 + 1, x0, x0 + 1 };
    int ys[4] = { y0, y0, y0 + 1, y0 + 1 };
#pragma unroll
    for (int cr = 0; cr < 4; cr++) {
      int xi = xs[cr], yi = ys[cr];
      if (xi >= 0 && xi < IMG_W && yi >= 0 && yi < IMG_H) {
        const float* vp = value + ((size_t)(yi * IMG_W + xi) * NHEAD + m) * DHEAD + d0;
        float wl = aw[p] * wgt[cr];
        a0 += wl * vp[0]; a1 += wl * vp[1]; a2 += wl * vp[2]; a3 += wl * vp[3];
      }
    }
  }
  half4v o;
  o[0] = (_Float16)a0; o[1] = (_Float16)a1; o[2] = (_Float16)a2; o[3] = (_Float16)a3;
  *(half4v*)(attn_s + (size_t)q * CDIM + m * DHEAD + d0) = o;
}

// ---------------- Multiscale depthwise 3x3 + exact GELU, f32 -> f16 ----------------
__global__ __launch_bounds__(256) void dwconv_gelu(const float* __restrict__ y,
    const float* __restrict__ wdw, const float* __restrict__ bdw,
    _Float16* __restrict__ out)
{
  int tid = blockIdx.x * 256 + threadIdx.x;
  int t = tid / HIDF;
  int c = tid - t * HIDF;
  int base, Hc, Wc, ti;
  if (t < 16384)      { base = 0;     Hc = 128; Wc = 128; ti = t; }
  else if (t < 20480) { base = 16384; Hc = 64;  Wc = 64;  ti = t - 16384; }
  else                { base = 20480; Hc = 32;  Wc = 32;  ti = t - 20480; }
  int h = ti / Wc, w = ti - (ti / Wc) * Wc;
  float acc = bdw[c];
#pragma unroll
  for (int kh = 0; kh < 3; kh++)
#pragma unroll
    for (int kw = 0; kw < 3; kw++) {
      int hh = h + kh - 1, ww = w + kw - 1;
      if (hh >= 0 && hh < Hc && ww >= 0 && ww < Wc)
        acc += y[(size_t)(base + hh * Wc + ww) * HIDF + c] * wdw[c * 9 + kh * 3 + kw];
    }
  float ge = 0.5f * acc * (1.0f + erff(acc * 0.70710678118654752f));
  out[(size_t)t * HIDF + c] = (_Float16)ge;
}

// ---------------- Weight convert+transpose: (K,N) f32 -> (Npad,K) f16, zero-pad ---
__global__ __launch_bounds__(256) void wconv_kernel(const float* __restrict__ W,
    _Float16* __restrict__ Wt, int K, int N, int Npad)
{
  int idx = blockIdx.x * 256 + threadIdx.x;
  if (idx >= Npad * K) return;
  int n = idx / K, k = idx - n * K;
  Wt[idx] = (_Float16)((n < N) ? W[(size_t)k * N + n] : 0.0f);
}

__global__ void pad_bias_kernel(const float* __restrict__ b, float* __restrict__ out,
                                int N, int Npad)
{
  int i = threadIdx.x;
  if (i < Npad) out[i] = (i < N) ? b[i] : 0.0f;
}

// ---------------- Host orchestration ----------------
extern "C" void kernel_launch(void* const* d_in, const int* in_sizes, int n_in,
                              void* d_out, int out_size, void* d_ws, size_t ws_size,
                              hipStream_t stream) {
  const float* query = (const float*)d_in[0];
  const float* refp  = (const float*)d_in[1];
  const float* feat  = (const float*)d_in[2];
  const float* gq = (const float*)d_in[3],  * bq = (const float*)d_in[4];
  const float* gf = (const float*)d_in[5],  * bf = (const float*)d_in[6];
  const float* W_off = (const float*)d_in[7],  * b_off = (const float*)d_in[8];
  const float* W_attn= (const float*)d_in[9],  * b_attn= (const float*)d_in[10];
  const float* W_val = (const float*)d_in[11], * b_val = (const float*)d_in[12];
  const float* W_out = (const float*)d_in[13], * b_out = (const float*)d_in[14];
  const float* gn = (const float*)d_in[15], * bn = (const float*)d_in[16];
  const float* W1 = (const float*)d_in[17], * b1 = (const float*)d_in[18];
  const float* Wdw= (const float*)d_in[19], * bdw= (const float*)d_in[20];
  const float* W2 = (const float*)d_in[21], * b2 = (const float*)d_in[22];
  float* out = (float*)d_out;

  char* ws = (char*)d_ws;
  size_t cur = 0;
  auto alloc = [&](size_t bytes) { size_t r = cur; cur = (cur + bytes + 255) & ~(size_t)255; return r; };

  _Float16* q_ln   = (_Float16*)(ws + alloc((size_t)LQ  * CDIM * 2)); // reused later as xn
  _Float16* f_ln   = (_Float16*)(ws + alloc((size_t)LIN * CDIM * 2));
  float*    value  = (float*)   (ws + alloc((size_t)LIN * CDIM * 4));
  float*    offb   = (float*)   (ws + alloc((size_t)LQ  * 96   * 4));
  float*    awl    = (float*)   (ws + alloc((size_t)LQ  * 32   * 4));
  _Float16* attn_s = (_Float16*)(ws + alloc((size_t)LQ  * CDIM * 2));
  float*    xbuf   = (float*)   (ws + alloc((size_t)LQ  * CDIM * 4));
  float*    ybuf   = (float*)   (ws + alloc((size_t)LQ  * HIDF * 4));
  _Float16* yact   = (_Float16*)(ws + alloc((size_t)LQ  * HIDF * 2));
  _Float16* Wt_val = (_Float16*)(ws + alloc((size_t)CDIM * CDIM * 2));
  _Float16* Wt_off = (_Float16*)(ws + alloc((size_t)96   * CDIM * 2));
  _Float16* Wt_atn = (_Float16*)(ws + alloc((size_t)32   * CDIM * 2));
  _Float16* Wt_out = (_Float16*)(ws + alloc((size_t)CDIM * CDIM * 2));
  _Float16* Wt1    = (_Float16*)(ws + alloc((size_t)HIDF * CDIM * 2));
  _Float16* Wt2    = (_Float16*)(ws + alloc((size_t)CDIM * HIDF * 2));
  float*    b_atnp = (float*)   (ws + alloc(32 * 4));
  _Float16* xn = q_ln;   // q_ln is dead by the time LN(x) runs

  auto cdiv = [](long a, long b) { return (unsigned)((a + b - 1) / b); };

  // weight transposes (f32 KxN -> f16 NpadxK)
  wconv_kernel<<<cdiv((long)CDIM*CDIM,256),256,0,stream>>>(W_val, Wt_val, CDIM, CDIM, CDIM);
  wconv_kernel<<<cdiv((long)96*CDIM,256),  256,0,stream>>>(W_off, Wt_off, CDIM, 96, 96);
  wconv_kernel<<<cdiv((long)32*CDIM,256),  256,0,stream>>>(W_attn, Wt_atn, CDIM, 24, 32);
  wconv_kernel<<<cdiv((long)CDIM*CDIM,256),256,0,stream>>>(W_out, Wt_out, CDIM, CDIM, CDIM);
  wconv_kernel<<<cdiv((long)HIDF*CDIM,256),256,0,stream>>>(W1, Wt1, CDIM, HIDF, HIDF);
  wconv_kernel<<<cdiv((long)CDIM*HIDF,256),256,0,stream>>>(W2, Wt2, HIDF, CDIM, CDIM);
  pad_bias_kernel<<<1,32,0,stream>>>(b_attn, b_atnp, 24, 32);

  // LayerNorms of query and feat -> f16
  ln_kernel<<<LQ, 256, 0, stream>>>(query, gq, bq, q_ln);
  ln_kernel<<<LIN, 256, 0, stream>>>(feat, gf, bf, f_ln);

  // value = f_ln @ W_val + b_val    (4096 x 768)
  gemm_wmma<<<dim3(CDIM/32, LIN/256), 256, 0, stream>>>(f_ln, Wt_val, b_val, nullptr, value, CDIM, CDIM);
  // off = q_ln @ W_off + b_off      (21504 x 96)
  gemm_wmma<<<dim3(96/32, LQ/256), 256, 0, stream>>>(q_ln, Wt_off, b_off, nullptr, offb, 96, CDIM);
  // attn logits = q_ln @ W_attn     (21504 x 32, cols 24..31 padded)
  gemm_wmma<<<dim3(32/32, LQ/256), 256, 0, stream>>>(q_ln, Wt_atn, b_atnp, nullptr, awl, 32, CDIM);

  // softmax + bilinear sampling -> attn_s f16 (21504 x 768)
  sample_kernel<<<(LQ * NHEAD) / 8, 256, 0, stream>>>(refp, offb, awl, value, attn_s);

  // x = query + attn_s @ W_out + b_out
  gemm_wmma<<<dim3(CDIM/32, LQ/256), 256, 0, stream>>>(attn_s, Wt_out, b_out, query, xbuf, CDIM, CDIM);

  // LN(x) -> xn f16 ; y = xn @ W1 + b1 (21504 x 192)
  ln_kernel<<<LQ, 256, 0, stream>>>(xbuf, gn, bn, xn);
  gemm_wmma<<<dim3(HIDF/32, LQ/256), 256, 0, stream>>>(xn, Wt1, b1, nullptr, ybuf, HIDF, CDIM);

  // multiscale depthwise 3x3 + GELU -> yact f16
  dwconv_gelu<<<cdiv((long)LQ * HIDF, 256), 256, 0, stream>>>(ybuf, Wdw, bdw, yact);

  // out = x + yact @ W2 + b2
  gemm_wmma<<<dim3(CDIM/32, LQ/256), 256, 0, stream>>>(yact, Wt2, b2, xbuf, out, CDIM, HIDF);
}
